// LocalSelfAttention_66941360275743
// MI455X (gfx1250) — matrix operand order repaired and checked by reading, not compiled, not run on hardware
//
#include <hip/hip_runtime.h>
#include <hip/hip_bf16.h>

typedef __bf16 bf16_t;
typedef __attribute__((ext_vector_type(16))) __bf16 v16bf;
typedef __attribute__((ext_vector_type(8)))  __bf16 v8bf;
typedef __attribute__((ext_vector_type(4)))  __bf16 v4bf;
typedef __attribute__((ext_vector_type(8)))  float  v8f;

#define SEQ   4096
#define EMB   1024
#define NTOK  16384L          // B*S = 4*4096
#define WND   256
#define BQ    64
#define SCALE_F 0.125f        // HEAD_DIM^-0.5 = 64^-0.5
#define KC    32
#define LDT   40              // padded LDS stride (bf16 elems) for 32-wide K chunks
#define BM    128
#define BN    128
#define PST   (WND + 8)       // padded stride of P buffer
#define BUFB  (BM * LDT * 2)  // bytes per LDS staging buffer (10240)

// async copy: 16B global -> LDS, tracked by ASYNCcnt (no VGPR data movement)
#define ASYNC_LD16(ldsoff, gptr)                                   \
  asm volatile("global_load_async_to_lds_b128 %0, %1, off"         \
               :: "v"(ldsoff), "v"(gptr) : "memory")

// ---------------------------------------------------------------- fragments
__device__ __forceinline__ v16bf make16(v8bf lo, v8bf hi) {
  v16bf f;
#pragma unroll
  for (int i = 0; i < 8; ++i) { f[i] = lo[i]; f[i + 8] = hi[i]; }
  return f;
}

// A fragment (16x32, MxK): lane l -> row (l&15); 16 contiguous K values at
// offset (l>>4)*16.  Source tile is row-major with stride ld.
__device__ __forceinline__ v16bf load_afrag(const bf16_t* tile, int row0,
                                            int ld, int koff) {
  int l = threadIdx.x & 31;
  const bf16_t* p = tile + (row0 + (l & 15)) * ld + koff + ((l >> 4) << 4);
  return make16(*(const v8bf*)p, *(const v8bf*)(p + 8));
}

// B fragment (32x16, KxN) read from an N-major (N x K row-major) tile:
// lane l -> column (l&15); 16 contiguous K values at offset (l>>4)*16.
__device__ __forceinline__ v16bf load_bfrag_lds(const bf16_t* tile, int row0,
                                                int ld, int koff) {
  int l = threadIdx.x & 31;
  const bf16_t* p = tile + (row0 + (l & 15)) * ld + koff + ((l >> 4) << 4);
  return make16(*(const v8bf*)p, *(const v8bf*)(p + 8));
}

__device__ __forceinline__ v16bf load_bfrag_g(const bf16_t* base, long ld) {
  int l = threadIdx.x & 31;
  const bf16_t* p = base + (long)(l & 15) * ld + ((l >> 4) << 4);
  return make16(*(const v8bf*)p, *(const v8bf*)(p + 8));
}

__device__ __forceinline__ v8f wmma_bf16(v16bf a, v16bf b, v8f c) {
  return __builtin_amdgcn_wmma_f32_16x16x32_bf16(false, a, false, b,
                                                 (short)0, c, false, false);
}

// ---------------------------------------------------------------- conversion
__global__ void cvt_f32_bf16(const float* __restrict__ in,
                             bf16_t* __restrict__ out, long n) {
  long i = 4L * ((long)blockIdx.x * blockDim.x + threadIdx.x);
  if (i < n) {
    float4 f = *(const float4*)(in + i);
    v4bf o;
    o[0] = (bf16_t)f.x; o[1] = (bf16_t)f.y;
    o[2] = (bf16_t)f.z; o[3] = (bf16_t)f.w;
    *(v4bf*)(out + i) = o;
  }
}

// ---------------------------------------------------------------- GEMM
// C[M=16384, N=1024] = A[M,K=1024] @ W[N,K]^T + bias
// OUT_MODE 0: bf16 row-major; 1: bf16 transposed [N,M]; 2: f32 + residual
// Double-buffered LDS staging via GLOBAL_LOAD_ASYNC_TO_LDS_B128 (ASYNCcnt):
// the next K chunk streams into the other buffer while WMMAs consume this one.
template <int OUT_MODE>
__global__ __launch_bounds__(256) void gemm_kernel(
    const bf16_t* __restrict__ A, const bf16_t* __restrict__ W,
    const float* __restrict__ bias, const float* __restrict__ resid,
    void* __restrict__ outv) {
  __shared__ bf16_t As[2][BM * LDT];   // 2 x 10 KB
  __shared__ bf16_t Ws[2][BN * LDT];   // 2 x 10 KB
  const int mB = blockIdx.y * BM;
  const int nB = blockIdx.x * BN;
  const int t  = threadIdx.x;
  const int w  = t >> 5;
  const int rg = w >> 1;      // 4 row groups of 32 rows
  const int ch = w & 1;       // 2 col halves of 64 cols

  v8f acc[2][4];
#pragma unroll
  for (int i = 0; i < 2; ++i)
#pragma unroll
    for (int j = 0; j < 4; ++j)
#pragma unroll
      for (int r = 0; r < 8; ++r) acc[i][j][r] = 0.f;

  // per-thread staging slot: row r, 16-elem half c
  const int r = t >> 1, c = (t & 1) * 16;
  const bf16_t* ga = A + (long)(mB + r) * EMB + c;
  const bf16_t* gw = W + (long)(nB + r) * EMB + c;
  // low 32 bits of a generic LDS pointer == LDS byte offset (ISA 10.2)
  const unsigned laA = (unsigned)(unsigned long long)&As[0][r * LDT + c];
  const unsigned laW = (unsigned)(unsigned long long)&Ws[0][r * LDT + c];

  // issue one chunk's 4 async 16B copies into buffer `buf`
  auto issue = [&](int buf, int k0) {
    unsigned a0 = laA + (unsigned)buf * BUFB;
    unsigned w0 = laW + (unsigned)buf * BUFB;
    ASYNC_LD16(a0,      ga + k0);
    ASYNC_LD16(a0 + 16, ga + k0 + 8);
    ASYNC_LD16(w0,      gw + k0);
    ASYNC_LD16(w0 + 16, gw + k0 + 8);
  };

  issue(0, 0);
  int cur = 0;
  for (int k0 = 0; k0 < EMB; k0 += KC) {
    asm volatile("s_wait_asynccnt 0x0" ::: "memory");  // my chunk landed
    __syncthreads();                                   // everyone's landed
    if (k0 + KC < EMB) issue(cur ^ 1, k0 + KC);        // stream next chunk
    const bf16_t* Ab = As[cur];
    const bf16_t* Wb = Ws[cur];
    v16bf a0 = load_afrag(Ab, rg * 32, LDT, 0);
    v16bf a1 = load_afrag(Ab, rg * 32 + 16, LDT, 0);
#pragma unroll
    for (int j = 0; j < 4; ++j) {
      v16bf b = load_bfrag_lds(Wb, ch * 64 + j * 16, LDT, 0);
      acc[0][j] = wmma_bf16(a0, b, acc[0][j]);
      acc[1][j] = wmma_bf16(a1, b, acc[1][j]);
    }
    cur ^= 1;
  }

  const int l  = t & 31;
  const int lm = (l >> 4) * 8;
  const int ln = l & 15;
#pragma unroll
  for (int i = 0; i < 2; ++i)
#pragma unroll
    for (int j = 0; j < 4; ++j) {
      int n = nB + ch * 64 + j * 16 + ln;
      float bv = bias[n];
      long mbase = mB + rg * 32 + i * 16 + lm;
      if (OUT_MODE == 1) {
        // transposed [N][M]: lane's 8 values are 8 consecutive m -> one 16B store
        v8bf pk;
#pragma unroll
        for (int rr = 0; rr < 8; ++rr) pk[rr] = (bf16_t)(acc[i][j][rr] + bv);
        *(v8bf*)&((bf16_t*)outv)[(long)n * NTOK + mbase] = pk;
      } else {
#pragma unroll
        for (int rr = 0; rr < 8; ++rr) {
          long m = mbase + rr;
          float v = acc[i][j][rr] + bv;
          if (OUT_MODE == 0)
            ((bf16_t*)outv)[m * EMB + n] = (bf16_t)v;
          else
            ((float*)outv)[m * EMB + n] = v + resid[m * EMB + n];
        }
      }
    }
}

// ---------------------------------------------------------------- attention
// One block: 64 q-rows of one window. 4 waves, wave w owns rows [w*16,w*16+16).
// Specialized on NCT (number of 16-wide score column tiles) so all tile loops
// are compile-time bounded -> no per-tile scalar branching.
template <int NCT>
__device__ __forceinline__ void attn_body(
    const bf16_t* __restrict__ q, const bf16_t* __restrict__ k,
    const bf16_t* __restrict__ vt, bf16_t* __restrict__ o,
    bf16_t* Qs, bf16_t* Ks, bf16_t* Ps, int wid, int w, int l, int t) {
  constexpr int QB    = NCT / 4 - 1;   // q sub-block within window
  constexpr int NCOLS = NCT * 16;      // causal column bound (64..256)
  const long tQ = (long)wid * WND + QB * BQ;
  const long tK = (long)wid * WND;

  // ---- S = Q @ K^T : wave holds 16 rows x NCOLS cols in registers
  v8f acc[NCT];
#pragma unroll
  for (int ct = 0; ct < NCT; ++ct)
#pragma unroll
    for (int rr = 0; rr < 8; ++rr) acc[ct][rr] = 0.f;

  for (int e0 = 0; e0 < EMB; e0 += KC) {
    __syncthreads();
    {
      int r = t >> 1, c = (t & 1) * 16;
      const bf16_t* sq = q + (tQ + r) * EMB + e0 + c;
      *(v8bf*)&Qs[r * LDT + c]     = *(const v8bf*)sq;
      *(v8bf*)&Qs[r * LDT + c + 8] = *(const v8bf*)(sq + 8);
#pragma unroll
      for (int rr = 0; rr < NCT / 4; ++rr) {   // stage only needed K rows
        int kr = (t >> 1) + rr * 64;
        const bf16_t* sk = k + (tK + kr) * EMB + e0 + c;
        *(v8bf*)&Ks[kr * LDT + c]     = *(const v8bf*)sk;
        *(v8bf*)&Ks[kr * LDT + c + 8] = *(const v8bf*)(sk + 8);
      }
    }
    __syncthreads();
    v16bf a = load_afrag(Qs, w * 16, LDT, 0);
#pragma unroll
    for (int ct = 0; ct < NCT; ++ct) {
      v16bf b = load_bfrag_lds(Ks, ct * 16, LDT, 0);
      acc[ct] = wmma_bf16(a, b, acc[ct]);
    }
  }

  // ---- causal mask + softmax in registers; write bf16 probs to LDS
  const int lanecol = l & 15;
  const int rowbase = w * 16 + ((l >> 4) << 3);   // local row of j=0
#pragma unroll
  for (int j = 0; j < 8; ++j) {
    const int rowpos = QB * BQ + rowbase + j;     // row within window
    float mx = -3.4e38f;
#pragma unroll
    for (int ct = 0; ct < NCT; ++ct) {
      int cc = ct * 16 + lanecol;
      float v = acc[ct][j] * SCALE_F;
      v = (cc <= rowpos) ? v : -3.4e38f;
      acc[ct][j] = v;
      mx = fmaxf(mx, v);
    }
#pragma unroll
    for (int mmask = 1; mmask < 16; mmask <<= 1)
      mx = fmaxf(mx, __shfl_xor(mx, mmask, 32));
    float sum = 0.f;
#pragma unroll
    for (int ct = 0; ct < NCT; ++ct) {
      float p = __expf(acc[ct][j] - mx);
      acc[ct][j] = p;
      sum += p;
    }
#pragma unroll
    for (int mmask = 1; mmask < 16; mmask <<= 1)
      sum += __shfl_xor(sum, mmask, 32);
    float inv = 1.f / sum;
#pragma unroll
    for (int ct = 0; ct < NCT; ++ct)
      Ps[(rowbase + j) * PST + ct * 16 + lanecol] =
          (bf16_t)(acc[ct][j] * inv);
  }
  __syncthreads();

  // ---- O = P @ V : 16 rows x 1024 cols, 128-col chunks, V^T rows contiguous
#pragma unroll 1
  for (int nc = 0; nc < 8; ++nc) {
    const int nB = nc * 128;
    v8f oacc[8];
#pragma unroll
    for (int j = 0; j < 8; ++j)
#pragma unroll
      for (int rr = 0; rr < 8; ++rr) oacc[j][rr] = 0.f;

#pragma unroll
    for (int kk = 0; kk < NCOLS; kk += KC) {
      v16bf a = load_afrag(Ps, w * 16, PST, kk);
#pragma unroll
      for (int j = 0; j < 8; ++j) {
        v16bf b = load_bfrag_g(vt + (long)(nB + j * 16) * NTOK + tK + kk, NTOK);
        oacc[j] = wmma_bf16(a, b, oacc[j]);
      }
    }
    const int lm = (l >> 4) * 8;
#pragma unroll
    for (int j = 0; j < 8; ++j) {
      long n = nB + j * 16 + (l & 15);
#pragma unroll
      for (int rr = 0; rr < 8; ++rr) {
        long m = tQ + w * 16 + lm + rr;
        o[m * EMB + n] = (bf16_t)oacc[j][rr];
      }
    }
  }
}

__global__ __launch_bounds__(128) void attn_kernel(
    const bf16_t* __restrict__ q, const bf16_t* __restrict__ k,
    const bf16_t* __restrict__ vt, bf16_t* __restrict__ o) {
  __shared__ bf16_t Qs[BQ * LDT];    //  5.0 KB
  __shared__ bf16_t Ks[WND * LDT];   // 20.0 KB
  __shared__ bf16_t Ps[BQ * PST];    // 33.0 KB   (total < 64 KB)
  const int qb  = blockIdx.x & 3;
  const int wid = blockIdx.x >> 2;   // window id 0..63 (b*16+nw)
  const int t = threadIdx.x;
  const int w = t >> 5;
  const int l = t & 31;
  switch (qb) {                      // uniform per-block branch
    case 0: attn_body<4>(q, k, vt, o, Qs, Ks, Ps, wid, w, l, t); break;
    case 1: attn_body<8 >(q, k, vt, o, Qs, Ks, Ps, wid, w, l, t); break;
    case 2: attn_body<12>(q, k, vt, o, Qs, Ks, Ps, wid, w, l, t); break;
    default: attn_body<16>(q, k, vt, o, Qs, Ks, Ps, wid, w, l, t); break;
  }
}

// ---------------------------------------------------------------- launch
extern "C" void kernel_launch(void* const* d_in, const int* in_sizes, int n_in,
                              void* d_out, int out_size, void* d_ws,
                              size_t ws_size, hipStream_t stream) {
  const float* x  = (const float*)d_in[0];
  const float* Wq = (const float*)d_in[1];
  const float* bq = (const float*)d_in[2];
  const float* Wk = (const float*)d_in[3];
  const float* bk = (const float*)d_in[4];
  const float* Wv = (const float*)d_in[5];
  const float* bv = (const float*)d_in[6];
  const float* Wo = (const float*)d_in[7];
  const float* bo = (const float*)d_in[8];
  float* out = (float*)d_out;

  bf16_t* xb  = (bf16_t*)d_ws;
  bf16_t* wqb = xb  + NTOK * EMB;
  bf16_t* wkb = wqb + (long)EMB * EMB;
  bf16_t* wvb = wkb + (long)EMB * EMB;
  bf16_t* wob = wvb + (long)EMB * EMB;
  bf16_t* qb_ = wob + (long)EMB * EMB;
  bf16_t* kb_ = qb_ + NTOK * EMB;
  bf16_t* vtb = kb_ + NTOK * EMB;   // stored transposed [EMB][NTOK]
  bf16_t* pb  = vtb + NTOK * EMB;   // attention output, row-major

  const long nX = NTOK * EMB;       // 16,777,216
  const long nW = (long)EMB * EMB;  //  1,048,576
  cvt_f32_bf16<<<(int)(nX / 4 / 256), 256, 0, stream>>>(x,  xb,  nX);
  cvt_f32_bf16<<<(int)(nW / 4 / 256), 256, 0, stream>>>(Wq, wqb, nW);
  cvt_f32_bf16<<<(int)(nW / 4 / 256), 256, 0, stream>>>(Wk, wkb, nW);
  cvt_f32_bf16<<<(int)(nW / 4 / 256), 256, 0, stream>>>(Wv, wvb, nW);
  cvt_f32_bf16<<<(int)(nW / 4 / 256), 256, 0, stream>>>(Wo, wob, nW);

  dim3 gg(EMB / BN, NTOK / BM);     // (8, 128)
  gemm_kernel<0><<<gg, 256, 0, stream>>>(xb, wqb, bq, nullptr, (void*)qb_);
  gemm_kernel<0><<<gg, 256, 0, stream>>>(xb, wkb, bk, nullptr, (void*)kb_);
  gemm_kernel<1><<<gg, 256, 0, stream>>>(xb, wvb, bv, nullptr, (void*)vtb);

  attn_kernel<<<256, 128, 0, stream>>>(qb_, kb_, vtb, pb);

  gemm_kernel<2><<<gg, 256, 0, stream>>>(pb, wob, bo, x, (void*)out);
}
